// MultiHeadAttention_15719580304405
// MI455X (gfx1250) — matrix operand order repaired and checked by reading, not compiled, not run on hardware
//
#include <hip/hip_runtime.h>
#include <math.h>

// ---------------------------------------------------------------------------
// MHA forward for MI455X (gfx1250), wave32 + WMMA bf16 16x16x32 + TDM.
// Pipeline:
//   1) one-shot f32 -> bf16 conversion of activations + weights (workspace)
//   2) QKV projections: pure-bf16 WMMA GEMMs (no conversion in inner loop)
//   3) flash attention: K and V tiles DMA'd to LDS via tensor_load_to_lds,
//      V transposed LDS->LDS, QK^T / PV via WMMA, exp2-domain online softmax
//   4) output projection (+bias) to f32
// ---------------------------------------------------------------------------

#define EMB   1024
#define NH    16
#define HD    64
#define SEQ   2048
#define BATCH 2
#define MTOT  (BATCH * SEQ)   // 4096 rows

typedef __attribute__((ext_vector_type(16))) __bf16 v16bf;
typedef __attribute__((ext_vector_type(8)))  float  v8f;
typedef __attribute__((ext_vector_type(4)))  unsigned int u32x4;
typedef __attribute__((ext_vector_type(8)))  int  i32x8;
typedef __attribute__((ext_vector_type(4)))  int  i32x4;

#if __has_builtin(__builtin_amdgcn_tensor_load_to_lds) && \
    __has_builtin(__builtin_amdgcn_s_wait_tensorcnt)
#define USE_TDM 1
#endif

union Frag16 {
  v16bf v;
  unsigned short u[16];
};

__device__ __forceinline__ unsigned short f2bf(float f) {
  // round-to-nearest-even f32 -> bf16 (bit-level)
  unsigned int x = __float_as_uint(f);
  unsigned int r = (x + 0x7fffu + ((x >> 16) & 1u)) >> 16;
  return (unsigned short)r;
}

__device__ __forceinline__ unsigned int pack2bf(float lo, float hi) {
  return (unsigned int)f2bf(lo) | ((unsigned int)f2bf(hi) << 16);
}

// ---------------------------------------------------------------------------
// One-shot f32 -> bf16 conversion (8 elements / thread, b128 in, b128 out)
// ---------------------------------------------------------------------------
__global__ __launch_bounds__(256)
void cvt_f32_bf16_kernel(const float* __restrict__ src,
                         unsigned short* __restrict__ dst, int n8) {
  const int idx = blockIdx.x * blockDim.x + threadIdx.x;
  if (idx >= n8) return;
  const float4* s = (const float4*)src + (size_t)idx * 2;
  const float4 f0 = s[0];
  const float4 f1 = s[1];
  uint4 p;
  p.x = pack2bf(f0.x, f0.y);
  p.y = pack2bf(f0.z, f0.w);
  p.z = pack2bf(f1.x, f1.y);
  p.w = pack2bf(f1.z, f1.w);
  ((uint4*)dst)[idx] = p;
}

// ---------------------------------------------------------------------------
// Pure-bf16 GEMM: out[m,n] = sum_k A[m,k] * W[n,k]  (+ optional bias, f32 out)
// Block = 256 threads (8 waves), tile 128(M) x 64(N), K step 32.
// A frag (16-bit A 16x32): lane (l16,hi) holds row l16,
//   element e -> K = 16*hi + e.
// B frag (16-bit B 32x16): lane holds col l16, element e -> K = 16*hi + e.
// C layout: VGPR j -> row j+8*hi, col l16.
// ---------------------------------------------------------------------------
__global__ __launch_bounds__(256)
void gemm_bf16_kernel(const unsigned short* __restrict__ A,
                      const unsigned short* __restrict__ W,
                      unsigned short* __restrict__ outBf,
                      float* __restrict__ outF,
                      const float* __restrict__ bias) {
  const int lane = threadIdx.x & 31;
  const int wave = threadIdx.x >> 5;
  const int hi   = lane >> 4;
  const int l16  = lane & 15;
  const int rowBase = blockIdx.y * 128 + wave * 16;
  const int colBase = blockIdx.x * 64;

  v8f acc[4];
#pragma unroll
  for (int t = 0; t < 4; ++t)
#pragma unroll
    for (int j = 0; j < 8; ++j) acc[t][j] = 0.0f;

  const unsigned short* arow = A + (size_t)(rowBase + l16) * EMB;

  for (int kb = 0; kb < EMB; kb += 32) {
    __builtin_prefetch(arow + kb + 256, 0, 1);   // global_prefetch_b8
    Frag16 a;
    const unsigned short* ap = arow + kb + 16 * hi;
#pragma unroll
    for (int i = 0; i < 8; ++i) { a.u[i] = ap[i]; a.u[8 + i] = ap[8 + i]; }
#pragma unroll
    for (int t = 0; t < 4; ++t) {
      Frag16 b;
      const unsigned short* bp =
          W + (size_t)(colBase + t * 16 + l16) * EMB + kb + 16 * hi;
#pragma unroll
      for (int i = 0; i < 16; ++i) b.u[i] = bp[i];
      acc[t] = __builtin_amdgcn_wmma_f32_16x16x32_bf16(
          false, a.v, false, b.v, (short)0, acc[t], false, false);
    }
  }

  if (outBf) {
#pragma unroll
    for (int t = 0; t < 4; ++t)
#pragma unroll
      for (int j = 0; j < 8; ++j) {
        const int r = rowBase + j + 8 * hi;
        const int c = colBase + t * 16 + l16;
        outBf[(size_t)r * EMB + c] = f2bf(acc[t][j]);
      }
  } else {
#pragma unroll
    for (int t = 0; t < 4; ++t)
#pragma unroll
      for (int j = 0; j < 8; ++j) {
        const int r = rowBase + j + 8 * hi;
        const int c = colBase + t * 16 + l16;
        outF[(size_t)r * EMB + c] = acc[t][j] + bias[c];
      }
  }
}

// ---------------------------------------------------------------------------
// Issue one 2D TDM tile load: 64x64 bf16 tile, row stride EMB, into LDS.
// Descriptor per ISA 8.3/8.4: group0 = {count=1, lds_addr, gaddr, type=2},
// group1 = {data_size=2B, tensor 64x64, tile 64x64, dim0_stride=EMB}.
// ---------------------------------------------------------------------------
#ifdef USE_TDM
__device__ __forceinline__ void tdm_tile_64x64(const unsigned short* gsrc,
                                               unsigned int ldsOff) {
  const unsigned long long ga = (unsigned long long)gsrc;
  u32x4 g0;
  g0[0] = 1u;
  g0[1] = ldsOff;
  g0[2] = (unsigned int)ga;
  g0[3] = ((unsigned int)((ga >> 32) & 0x01FFFFFFull)) | (2u << 30);
  i32x8 g1;
  g1[0] = (int)(1u << 16);    // data_size = 2 bytes
  g1[1] = (int)(64u << 16);   // tensor_dim0[15:0]
  g1[2] = (int)(64u << 16);   // tensor_dim0 hi | tensor_dim1[15:0]
  g1[3] = (int)(64u << 16);   // tensor_dim1 hi | tile_dim0
  g1[4] = (int)64;            // tile_dim1 (tile_dim2 = 0)
  g1[5] = (int)EMB;           // tensor_dim0_stride lo
  g1[6] = 0;
  g1[7] = 0;
  i32x4 z4 = {0, 0, 0, 0};
#if __clang_major__ >= 23
  i32x8 z8 = {0, 0, 0, 0, 0, 0, 0, 0};
  __builtin_amdgcn_tensor_load_to_lds(g0, g1, z4, z4, z8, 0);
#else
  __builtin_amdgcn_tensor_load_to_lds(g0, g1, z4, z4, 0);
#endif
}
#endif

// ---------------------------------------------------------------------------
// Flash attention. Grid: (S/64 q-tiles, B*NH). Block = 128 threads (4 waves),
// wave w owns q rows [64*qt + 16w, +16). Per 64-wide key tile:
//   K tile + V tile DMA'd to LDS by TDM (wave 0, s_wait_tensorcnt),
//   V transposed LDS->LDS, S = Q K^T (WMMA), scale+mask,
//   exp2-domain online softmax, P via LDS -> A operand, O += P V (WMMA).
// ---------------------------------------------------------------------------
__global__ __launch_bounds__(128)
void attn_kernel(const unsigned short* __restrict__ Qb,
                 const unsigned short* __restrict__ Kb,
                 const unsigned short* __restrict__ Vb,
                 const unsigned char* __restrict__ posmask,
                 const int* __restrict__ future_mask,
                 unsigned short* __restrict__ Ob) {
  const int lane = threadIdx.x & 31;
  const int wave = threadIdx.x >> 5;      // 0..3
  const int hi   = lane >> 4;
  const int l16  = lane & 15;
  const int qt   = blockIdx.x;            // 0..31
  const int bh   = blockIdx.y;            // 0..31
  const int b    = bh >> 4;
  const int h    = bh & 15;
  const bool causal = (future_mask[0] != 0);
  // scale * log2(e): softmax evaluated in exp2 domain (single v_exp_f32/elem)
  const float kScale = 0.125f * 1.44269504088896f;

  __shared__ unsigned short Klds[64 * HD];   // [n][d], K tile (TDM target)
  __shared__ unsigned short Vraw[64 * HD];   // [n][d], V tile (TDM target)
  __shared__ unsigned short Vlds[HD * 64];   // [d][n], V tile transposed
  __shared__ unsigned short Plds[64 * 64];   // [m][n], probs (bf16)

  // Resident Q fragments (K-dim = HD = 64 -> two 32-wide steps)
  Frag16 aq[2];
  const unsigned short* qrow =
      Qb + (size_t)(b * SEQ + qt * 64 + wave * 16 + l16) * EMB + h * HD;
#pragma unroll
  for (int ks = 0; ks < 2; ++ks) {
    const unsigned short* p = qrow + ks * 32 + 16 * hi;
#pragma unroll
    for (int i = 0; i < 8; ++i) { aq[ks].u[i] = p[i]; aq[ks].u[8 + i] = p[8 + i]; }
  }

  float mrow[8], lrow[8];
  v8f o[4];
#pragma unroll
  for (int j = 0; j < 8; ++j) { mrow[j] = -1e30f; lrow[j] = 0.0f; }
#pragma unroll
  for (int t = 0; t < 4; ++t)
#pragma unroll
    for (int j = 0; j < 8; ++j) o[t][j] = 0.0f;

  const int ktEnd = causal ? qt : (SEQ / 64 - 1);
  for (int kt = 0; kt <= ktEnd; ++kt) {
    __syncthreads();  // previous-iteration LDS readers are done

    // ---- stage K tile -> Klds[n][d], V tile -> Vraw[n][d] ----
#ifdef USE_TDM
    if (wave == 0) {
      const unsigned int kOff = (unsigned int)(unsigned long long)
          (__attribute__((address_space(3))) unsigned short*)Klds;
      const unsigned int vOff = (unsigned int)(unsigned long long)
          (__attribute__((address_space(3))) unsigned short*)Vraw;
      tdm_tile_64x64(Kb + (size_t)(b * SEQ + kt * 64) * EMB + h * HD, kOff);
      tdm_tile_64x64(Vb + (size_t)(b * SEQ + kt * 64) * EMB + h * HD, vOff);
      __builtin_amdgcn_s_wait_tensorcnt(0);
    }
#else
    for (int idx = threadIdx.x; idx < 64 * 8; idx += 128) {
      const int n  = idx >> 3;
      const int d0 = (idx & 7) * 8;
      const unsigned short* kp =
          Kb + (size_t)(b * SEQ + kt * 64 + n) * EMB + h * HD + d0;
      const unsigned short* vp =
          Vb + (size_t)(b * SEQ + kt * 64 + n) * EMB + h * HD + d0;
#pragma unroll
      for (int i = 0; i < 8; ++i) Klds[n * 64 + d0 + i] = kp[i];
#pragma unroll
      for (int i = 0; i < 8; ++i) Vraw[n * 64 + d0 + i] = vp[i];
    }
#endif
    __syncthreads();

    // ---- transpose V in LDS: Vlds[d][n] = Vraw[n][d] ----
    for (int idx = threadIdx.x; idx < 64 * 8; idx += 128) {
      const int n  = idx >> 3;
      const int d0 = (idx & 7) * 8;
      const unsigned short* vr = &Vraw[n * 64 + d0];   // b128 ds_load
#pragma unroll
      for (int i = 0; i < 8; ++i) Vlds[(d0 + i) * 64 + n] = vr[i];
    }
    __syncthreads();

    // ---- scores: S = Q @ K^T (16 x 64 per wave), K from LDS ----
    v8f s[4];
#pragma unroll
    for (int t = 0; t < 4; ++t)
#pragma unroll
      for (int j = 0; j < 8; ++j) s[t][j] = 0.0f;

#pragma unroll
    for (int t = 0; t < 4; ++t) {
#pragma unroll
      for (int ks = 0; ks < 2; ++ks) {
        Frag16 bk;  // B operand: col n = key row, elements = contiguous d
        const unsigned short* kp = &Klds[(t * 16 + l16) * 64 + ks * 32 + 16 * hi];
#pragma unroll
        for (int i = 0; i < 16; ++i) bk.u[i] = kp[i];
        s[t] = __builtin_amdgcn_wmma_f32_16x16x32_bf16(
            false, aq[ks].v, false, bk.v, (short)0, s[t], false, false);
      }
    }

    // key-position additive mask, hoisted out of the row loop
    float maskAdd[4];
#pragma unroll
    for (int t = 0; t < 4; ++t) {
      const int kg = kt * 64 + t * 16 + l16;
      maskAdd[t] = posmask[b * SEQ + kg] ? 0.0f : -1e30f;
    }

    // ---- exp2-domain online softmax ----
#pragma unroll
    for (int j = 0; j < 8; ++j) {
      const int qg = qt * 64 + wave * 16 + j + 8 * hi;
      float mx = -1e30f;
#pragma unroll
      for (int t = 0; t < 4; ++t) {
        const int kg = kt * 64 + t * 16 + l16;
        float v = s[t][j] * kScale + maskAdd[t];
        if (causal && kg > qg) v = -1e30f;
        s[t][j] = v;
        mx = fmaxf(mx, v);
      }
#pragma unroll
      for (int off = 1; off < 16; off <<= 1)        // reduce across 16-lane half
        mx = fmaxf(mx, __shfl_xor(mx, off, 32));
      const float mnew  = fmaxf(mrow[j], mx);
      const float alpha = __builtin_amdgcn_exp2f(mrow[j] - mnew);
      mrow[j] = mnew;
      float rs = 0.0f;
#pragma unroll
      for (int t = 0; t < 4; ++t) {
        const float p = __builtin_amdgcn_exp2f(s[t][j] - mnew);
        s[t][j] = p;
        rs += p;
      }
#pragma unroll
      for (int off = 1; off < 16; off <<= 1)
        rs += __shfl_xor(rs, off, 32);
      lrow[j] = lrow[j] * alpha + rs;
#pragma unroll
      for (int t = 0; t < 4; ++t) o[t][j] *= alpha;
    }

    // ---- P -> LDS (C layout -> memory), reload as A operand ----
#pragma unroll
    for (int t = 0; t < 4; ++t)
#pragma unroll
      for (int j = 0; j < 8; ++j)
        Plds[(wave * 16 + j + 8 * hi) * 64 + t * 16 + l16] = f2bf(s[t][j]);
    // own-wave rows only; same-wave LDS ops are in-order

    Frag16 apf[2];
#pragma unroll
    for (int ks = 0; ks < 2; ++ks) {
      const unsigned short* pp = &Plds[(wave * 16 + l16) * 64 + ks * 32 + 16 * hi];
#pragma unroll
      for (int i = 0; i < 8; ++i) { apf[ks].u[i] = pp[i]; apf[ks].u[8 + i] = pp[8 + i]; }
    }

    // ---- O += P @ V ----
#pragma unroll
    for (int t = 0; t < 4; ++t) {
#pragma unroll
      for (int ks = 0; ks < 2; ++ks) {
        Frag16 bv;  // B operand: col d = t*16+l16, K = key index n
        const unsigned short* vp = &Vlds[(t * 16 + l16) * 64 + ks * 32 + 16 * hi];
#pragma unroll
        for (int i = 0; i < 16; ++i) bv.u[i] = vp[i];
        o[t] = __builtin_amdgcn_wmma_f32_16x16x32_bf16(
            false, apf[ks].v, false, bv.v, (short)0, o[t], false, false);
      }
    }
  }

  // ---- epilogue: normalize, store bf16 O in (B,S,H*D) ----
#pragma unroll
  for (int t = 0; t < 4; ++t)
#pragma unroll
    for (int j = 0; j < 8; ++j) {
      const float val = o[t][j] / lrow[j];
      const int srow = b * SEQ + qt * 64 + wave * 16 + j + 8 * hi;
      Ob[(size_t)srow * EMB + h * HD + t * 16 + l16] = f2bf(val);
    }
}

// ---------------------------------------------------------------------------
extern "C" void kernel_launch(void* const* d_in, const int* in_sizes, int n_in,
                              void* d_out, int out_size, void* d_ws, size_t ws_size,
                              hipStream_t stream) {
  const float*         query = (const float*)d_in[0];
  const float*         key   = (const float*)d_in[1];
  const float*         value = (const float*)d_in[2];
  const unsigned char* pmask = (const unsigned char*)d_in[3];
  const int*           fmask = (const int*)d_in[4];
  const float*         Wq    = (const float*)d_in[5];
  const float*         Wk    = (const float*)d_in[6];
  const float*         Wv    = (const float*)d_in[7];
  const float*         Wo    = (const float*)d_in[8];
  const float*         bo    = (const float*)d_in[9];
  float*               out   = (float*)d_out;

  unsigned short* ws = (unsigned short*)d_ws;
  const size_t T = (size_t)MTOT * EMB;      // 4096*1024 elements
  const size_t Wsz = (size_t)EMB * EMB;     // 1024*1024 elements
  unsigned short* Xq  = ws;                 // bf16 activations
  unsigned short* Xk  = ws + 1 * T;
  unsigned short* Xv  = ws + 2 * T;
  unsigned short* Qb  = ws + 3 * T;         // bf16 projections
  unsigned short* Kb  = ws + 4 * T;
  unsigned short* Vb  = ws + 5 * T;
  unsigned short* Obf = ws + 6 * T;         // attention output (bf16)
  unsigned short* Wqb = ws + 7 * T;         // bf16 weights
  unsigned short* Wkb = Wqb + 1 * Wsz;
  unsigned short* Wvb = Wqb + 2 * Wsz;
  unsigned short* Wob = Wqb + 3 * Wsz;

  // 1) one-shot conversions
  const int actN8 = (int)(T / 8);
  const int wN8   = (int)(Wsz / 8);
  cvt_f32_bf16_kernel<<<(actN8 + 255) / 256, 256, 0, stream>>>(query, Xq, actN8);
  cvt_f32_bf16_kernel<<<(actN8 + 255) / 256, 256, 0, stream>>>(key,   Xk, actN8);
  cvt_f32_bf16_kernel<<<(actN8 + 255) / 256, 256, 0, stream>>>(value, Xv, actN8);
  cvt_f32_bf16_kernel<<<(wN8 + 255) / 256, 256, 0, stream>>>(Wq, Wqb, wN8);
  cvt_f32_bf16_kernel<<<(wN8 + 255) / 256, 256, 0, stream>>>(Wk, Wkb, wN8);
  cvt_f32_bf16_kernel<<<(wN8 + 255) / 256, 256, 0, stream>>>(Wv, Wvb, wN8);
  cvt_f32_bf16_kernel<<<(wN8 + 255) / 256, 256, 0, stream>>>(Wo, Wob, wN8);

  // 2) QKV projections (bf16 in, bf16 out)
  dim3 gemmGrid(EMB / 64, MTOT / 128);   // 16 x 32
  gemm_bf16_kernel<<<gemmGrid, 256, 0, stream>>>(Xq, Wqb, Qb, nullptr, nullptr);
  gemm_bf16_kernel<<<gemmGrid, 256, 0, stream>>>(Xk, Wkb, Kb, nullptr, nullptr);
  gemm_bf16_kernel<<<gemmGrid, 256, 0, stream>>>(Xv, Wvb, Vb, nullptr, nullptr);

  // 3) flash attention
  dim3 attnGrid(SEQ / 64, BATCH * NH);   // 32 x 32
  attn_kernel<<<attnGrid, 128, 0, stream>>>(Qb, Kb, Vb, pmask, fmask, Obf);

  // 4) output projection (+bias, f32 out)
  gemm_bf16_kernel<<<gemmGrid, 256, 0, stream>>>(Obf, Wob, nullptr, out, bo);
}
